// segSrCaps_17197049053609
// MI455X (gfx1250) — matrix operand order repaired and checked by reading, not compiled, not run on hardware
//
#include <hip/hip_runtime.h>
#include <hip/hip_bf16.h>
#include <math.h>

typedef __attribute__((ext_vector_type(16))) _Float16 v16h;
typedef __attribute__((ext_vector_type(8)))  _Float16 v8h;
typedef __attribute__((ext_vector_type(8)))  float    v8f;

#define WMMA_F16(A, B, C) \
  __builtin_amdgcn_wmma_f32_16x16x32_f16(false, (A), false, (B), (short)0, (C), false, false)

// Padded NHWC geometry (2-halo on each side; +2 extra rows for the K-pad tap in caps1)
#define C1_HP 134
#define C1_WP 132
#define V1_HP 132
#define V1_WP 132

// A-fragment (16x32 f16) per-lane element -> local K index (contiguous order):
// K = 16*half + e
__device__ __forceinline__ int a_kloc(int e, int half) {
  return (half << 4) + e;
}

// --------------------------------------------------------------------- utils
__global__ void fill_u32_kernel(unsigned* __restrict__ p, unsigned val, int n) {
  int idx = blockIdx.x * blockDim.x + threadIdx.x;
  if (idx < n) p[idx] = val;
}

// ---------------------------------------------------------------- small convs
__global__ void conv3_relu_kernel(const float* __restrict__ in, const float* __restrict__ w,
                                  float* __restrict__ out, int N, int Cin, int Cout) {
  int idx = blockIdx.x * blockDim.x + threadIdx.x;
  int total = N * Cout * 1024;
  if (idx >= total) return;
  int x  = idx & 31, yy = (idx >> 5) & 31;
  int co = (idx >> 10) % Cout, n = (idx >> 10) / Cout;
  float acc = 0.f;
  for (int ci = 0; ci < Cin; ++ci) {
    const float* ip = in + ((size_t)(n * Cin + ci) << 10);
    const float* wp = w + (size_t)(co * Cin + ci) * 9;
#pragma unroll
    for (int ky = 0; ky < 3; ++ky) {
      int iy = yy + ky - 1;
      if ((unsigned)iy >= 32u) continue;
#pragma unroll
      for (int kx = 0; kx < 3; ++kx) {
        int ix = x + kx - 1;
        if ((unsigned)ix >= 32u) continue;
        acc += ip[(iy << 5) + ix] * wp[ky * 3 + kx];
      }
    }
  }
  out[idx] = fmaxf(acc, 0.f);
}

// h(4,16,32,32) -> y(4,1,128,128): y[n, hh*4+a, ww*4+b] = h[n, a*4+b, hh, ww]; relu
__global__ void pixel_shuffle_relu_kernel(const float* __restrict__ h5, float* __restrict__ y) {
  int idx = blockIdx.x * blockDim.x + threadIdx.x;
  if (idx >= 4 * 128 * 128) return;
  int X = idx & 127, Y = (idx >> 7) & 127, n = idx >> 14;
  int hh = Y >> 2, a = Y & 3, ww = X >> 2, b = X & 3;
  float v = h5[(((size_t)n * 16 + (a * 4 + b)) * 32 + hh) * 32 + ww];
  y[idx] = fmaxf(v, 0.f);
}

// c1 = conv5x5(y, w_c1(16,1,5,5), pad 2) -> padded NHWC f16 [4][134][132][16], origin (2,2)
__global__ void conv5_c1_kernel(const float* __restrict__ y, const float* __restrict__ w_c1,
                                _Float16* __restrict__ c1h) {
  int idx = blockIdx.x * blockDim.x + threadIdx.x;
  if (idx >= 4 * 128 * 128) return;
  int w = idx & 127, h = (idx >> 7) & 127, n = idx >> 14;
  float patch[25];
#pragma unroll
  for (int q = 0; q < 25; ++q) {
    int hy = h + q / 5 - 2, wx = w + q % 5 - 2;
    patch[q] = ((unsigned)hy < 128u && (unsigned)wx < 128u)
                   ? y[((size_t)n * 128 + hy) * 128 + wx] : 0.f;
  }
  _Float16* outp = c1h + (((size_t)n * C1_HP + (h + 2)) * C1_WP + (w + 2)) * 16;
  for (int co = 0; co < 16; ++co) {
    float acc = 0.f;
#pragma unroll
    for (int q = 0; q < 25; ++q) acc += patch[q] * w_c1[co * 25 + q];
    outp[co] = (_Float16)acc;
  }
}

// ------------------------------------------------------------- weight packing
// A1: w_caps1(512,16,5,5) -> fragment order [cotile(32)][ks(13)][lane(32)][e(16)]
// K order: k = q*16 + ci  (q = ky*5+kx); rows with q>=25 are zero
__global__ void pack_a1_kernel(const float* __restrict__ w_caps1, _Float16* __restrict__ a1) {
  int idx = blockIdx.x * blockDim.x + threadIdx.x;
  if (idx >= 32 * 13 * 32 * 16) return;
  int e = idx & 15, lane = (idx >> 4) & 31, rest = idx >> 9;
  int ks = rest % 13, cotile = rest / 13;
  int m = lane & 15, half = lane >> 4;
  int k = ks * 32 + a_kloc(e, half);
  int ci = k & 15, q = k >> 4;
  float val = 0.f;
  if (q < 25) val = w_caps1[((size_t)(cotile * 16 + m) * 16 + ci) * 25 + q];
  a1[idx] = (_Float16)val;
}

// A2: w_caps2(1600,8,5,5) collapsed to dense (25->32, K=25*512)
// K order: k = q*512 + c, c = t*8+ci; W[z][c] = w_caps2[t*25+z][ci]
__global__ void pack_a2_kernel(const float* __restrict__ w_caps2, _Float16* __restrict__ a2) {
  int idx = blockIdx.x * blockDim.x + threadIdx.x;
  if (idx >= 2 * 400 * 32 * 16) return;
  int e = idx & 15, lane = (idx >> 4) & 31, rest = idx >> 9;
  int ks = rest % 400, ct = rest / 400;
  int m = lane & 15, half = lane >> 4;
  int k = ks * 32 + a_kloc(e, half);
  int q = k >> 9, c = k & 511;
  int t = c >> 3, ci = c & 7;
  int z = ct * 16 + m;
  float val = (z < 25) ? w_caps2[(((size_t)(t * 25 + z)) * 8 + ci) * 25 + q] : 0.f;
  a2[idx] = (_Float16)val;
}

// ----------------------------------------------------------- caps1 WMMA GEMM
// op[co, nhw] = W1(512x416) x im2col(c1h_pad); v1 = squash(op/64) over z-groups of 8.
// grid.x = 1024 strips (n,h,w0); block 256 = 8 waves; wave: 4 co-tiles x 4 N-tiles.
// All B loads are unconditional (halo-padded input); the q=25 K-pad tap multiplies zero A rows.
__global__ void __launch_bounds__(256) caps1_wmma_kernel(
    const _Float16* __restrict__ c1h, const _Float16* __restrict__ a1,
    _Float16* __restrict__ v1h) {
  const int strip = blockIdx.x;
  const int w0 = (strip & 1) << 6;
  const int h  = (strip >> 1) & 127;
  const int n  = strip >> 8;
  const int wave = threadIdx.x >> 5;
  const int lane = threadIdx.x & 31;
  const int half = lane >> 4;
  const int jlo  = lane & 15;

  v8f zacc = {};
  v8f acc[4][4];
#pragma unroll
  for (int a = 0; a < 4; ++a)
#pragma unroll
    for (int b = 0; b < 4; ++b) acc[a][b] = zacc;

  // Base pointers: B per N-tile (tap offset added per step), A per wave (imm offsets)
  const _Float16* pB[4];
#pragma unroll
  for (int ns = 0; ns < 4; ++ns)
    pB[ns] = c1h + (((size_t)n * C1_HP + h) * C1_WP + (w0 + ns * 16 + jlo)) * 16;
  const _Float16* pA = a1 + ((size_t)(wave * 4) * 13 * 32 + lane) * 16;

#pragma unroll
  for (int ks = 0; ks < 13; ++ks) {
    const int q = 2 * ks + half;                 // select between two consts per step
    const int toff = ((q / 5) * C1_WP + (q % 5)) * 16;
    v16h bfrag[4];
#pragma unroll
    for (int ns = 0; ns < 4; ++ns)
      bfrag[ns] = *(const v16h*)(pB[ns] + toff); // 32B contiguous, no predication
#pragma unroll
    for (int ct = 0; ct < 4; ++ct) {
      const v16h a = *(const v16h*)(pA + (ct * 13 + ks) * 512); // imm-offset b128 pair
#pragma unroll
      for (int ns = 0; ns < 4; ++ns)
        acc[ct][ns] = WMMA_F16(a, bfrag[ns], acc[ct][ns]);
    }
  }

  // Epilogue: the 8 acc VGPRs of a lane are the 8 z-values of one capsule at one pixel.
#pragma unroll
  for (int ct = 0; ct < 4; ++ct) {
    const int cobase = (wave * 4 + ct) * 16 + (half << 3);
#pragma unroll
    for (int ns = 0; ns < 4; ++ns) {
      float s[8];
      float sq = 0.f;
#pragma unroll
      for (int r = 0; r < 8; ++r) { s[r] = acc[ct][ns][r] * 0.015625f; sq += s[r] * s[r]; }
      const float sc = sq / (1.f + sq) * rsqrtf(sq + 1e-9f);
      v8h o;
#pragma unroll
      for (int r = 0; r < 8; ++r) o[r] = (_Float16)(s[r] * sc);
      const int w = w0 + ns * 16 + jlo;
      *(v8h*)(v1h + (((size_t)n * V1_HP + (h + 2)) * V1_WP + (w + 2)) * 512 + cobase) = o;
    }
  }
}

// ----------------------------------------------------------- caps2 WMMA GEMM
// s[z, nhw] = W2(32x12800) x im2col(v1h_pad); grid.x = 512 rows; 8 waves: 2 M x 8 N tiles.
// Tap-outer / channel-chunk-inner: all loads are immediate-offset b128, no predication.
__global__ void __launch_bounds__(256) caps2_wmma_kernel(
    const _Float16* __restrict__ v1h, const _Float16* __restrict__ a2,
    float* __restrict__ s2) {
  const int row = blockIdx.x;              // n*128 + h
  const int n = row >> 7, h = row & 127;
  const int wave = threadIdx.x >> 5, lane = threadIdx.x & 31;
  const int half = lane >> 4, jlo = lane & 15;
  const int ct = wave & 1;                 // M tile (z 0-15 / 16-31)
  const int j0 = ((wave >> 1) << 5) + jlo; // two N tiles: j0, j0+16
  v8f zacc = {};
  v8f acc0 = zacc, acc1 = zacc;
  const int hc = half << 4;

  const _Float16* pAb = a2 + ((size_t)ct * 400 * 32 + lane) * 16;

#pragma unroll 1
  for (int q = 0; q < 25; ++q) {
    const int ky = q / 5, kx = q % 5;      // amortized over 48 loads + 32 WMMAs
    const _Float16* pb0 =
        v1h + (((size_t)n * V1_HP + (h + ky)) * V1_WP + (j0 + kx)) * 512 + hc;
    const _Float16* pb1 = pb0 + 16 * 512;
    const _Float16* pa  = pAb + (size_t)q * 16 * 512;
    if (q + 1 < 25)
      __builtin_prefetch((const void*)(pa + 16 * 512), 0, 0);
#pragma unroll
    for (int cc = 0; cc < 16; ++cc) {
      const v16h a  = *(const v16h*)(pa + cc * 512);   // imm offsets, clause-friendly
      const v16h b0 = *(const v16h*)(pb0 + cc * 32);
      const v16h b1 = *(const v16h*)(pb1 + cc * 32);
      acc0 = WMMA_F16(a, b0, acc0);
      acc1 = WMMA_F16(a, b1, acc1);
    }
  }

  const size_t nb = (size_t)n * 128;
  const int zb = ct * 16 + (half << 3);
#pragma unroll
  for (int r = 0; r < 8; ++r) {
    const int z = zb + r;
    if (z < 25) {
      s2[((nb + h) * 128 + j0) * 25 + z]      = acc0[r];
      s2[((nb + h) * 128 + j0 + 16) * 25 + z] = acc1[r];
    }
  }
}

// squash over the 25 channels (NHWC)
__global__ void squash25_kernel(const float* __restrict__ s2, float* __restrict__ v2) {
  int idx = blockIdx.x * blockDim.x + threadIdx.x;
  if (idx >= 4 * 128 * 128) return;
  const float* s = s2 + (size_t)idx * 25;
  float sq = 0.f;
#pragma unroll
  for (int m = 0; m < 25; ++m) sq += s[m] * s[m];
  const float sc = sq / (1.f + sq) * rsqrtf(sq + 1e-9f);
#pragma unroll
  for (int m = 0; m < 25; ++m) v2[(size_t)idx * 25 + m] = s[m] * sc;
}

// out[n,h,w] = sum_m v2[n,h,w,m] * y[n, h+m%5-2, w+m/5-2] + y[n,h,w]
__global__ void pixel_conv_add_kernel(const float* __restrict__ v2, const float* __restrict__ y,
                                      float* __restrict__ out) {
  int idx = blockIdx.x * blockDim.x + threadIdx.x;
  if (idx >= 4 * 128 * 128) return;
  int w = idx & 127, h = (idx >> 7) & 127, n = idx >> 14;
  const float* kv = v2 + (size_t)idx * 25;
  float acc = 0.f;
#pragma unroll
  for (int m = 0; m < 25; ++m) {
    int hy = h + m % 5 - 2, wx = w + m / 5 - 2;
    float yv = ((unsigned)hy < 128u && (unsigned)wx < 128u)
                   ? y[((size_t)n * 128 + hy) * 128 + wx] : 0.f;
    acc += kv[m] * yv;
  }
  out[idx] = acc + y[idx];
}

// ------------------------------------------------------------------- launcher
extern "C" void kernel_launch(void* const* d_in, const int* in_sizes, int n_in,
                              void* d_out, int out_size, void* d_ws, size_t ws_size,
                              hipStream_t stream) {
  const float* x       = (const float*)d_in[0];
  const float* w_e1    = (const float*)d_in[1];
  const float* w_e2    = (const float*)d_in[2];
  const float* w_e3    = (const float*)d_in[3];
  const float* w_d1    = (const float*)d_in[4];
  const float* w_d2    = (const float*)d_in[5];
  const float* w_c1    = (const float*)d_in[6];
  const float* w_caps1 = (const float*)d_in[7];
  const float* w_caps2 = (const float*)d_in[8];

  char* wsp = (char*)d_ws;
  auto take = [&](size_t bytes) -> void* {
    void* p = (void*)wsp;
    wsp += (bytes + 255) & ~(size_t)255;
    return p;
  };
  const size_t c1h_elems = (size_t)4 * C1_HP * C1_WP * 16;   // padded NHWC f16
  const size_t v1h_elems = (size_t)4 * V1_HP * V1_WP * 512;  // padded NHWC f16

  float*    h1  = (float*)take((size_t)4 * 16 * 1024 * 4);
  float*    h2  = (float*)take((size_t)4 * 32 * 1024 * 4);
  float*    h3  = (float*)take((size_t)4 * 64 * 1024 * 4);
  float*    h4  = (float*)take((size_t)4 * 32 * 1024 * 4);
  float*    h5  = (float*)take((size_t)4 * 16 * 1024 * 4);
  float*    y   = (float*)take((size_t)4 * 128 * 128 * 4);
  _Float16* c1h = (_Float16*)take(c1h_elems * 2);
  _Float16* a1  = (_Float16*)take((size_t)32 * 13 * 32 * 16 * 2);
  _Float16* a2  = (_Float16*)take((size_t)2 * 400 * 32 * 16 * 2);
  _Float16* v1h = (_Float16*)take(v1h_elems * 2);
  float*    s2  = (float*)take((size_t)4 * 128 * 128 * 25 * 4);
  float*    v2  = (float*)take((size_t)4 * 128 * 128 * 25 * 4);

  const int B = 256;
  auto grid = [](size_t total) { return (int)((total + 255) / 256); };

  // zero halos (whole padded buffers; cheap at 23.3 TB/s)
  fill_u32_kernel<<<grid(c1h_elems / 2), B, 0, stream>>>((unsigned*)c1h, 0u, (int)(c1h_elems / 2));
  fill_u32_kernel<<<grid(v1h_elems / 2), B, 0, stream>>>((unsigned*)v1h, 0u, (int)(v1h_elems / 2));

  conv3_relu_kernel<<<grid(4 * 16 * 1024), B, 0, stream>>>(x,  w_e1, h1, 4, 1,  16);
  conv3_relu_kernel<<<grid(4 * 32 * 1024), B, 0, stream>>>(h1, w_e2, h2, 4, 16, 32);
  conv3_relu_kernel<<<grid(4 * 64 * 1024), B, 0, stream>>>(h2, w_e3, h3, 4, 32, 64);
  conv3_relu_kernel<<<grid(4 * 32 * 1024), B, 0, stream>>>(h3, w_d1, h4, 4, 64, 32);
  conv3_relu_kernel<<<grid(4 * 16 * 1024), B, 0, stream>>>(h4, w_d2, h5, 4, 32, 16);

  pixel_shuffle_relu_kernel<<<grid(4 * 128 * 128), B, 0, stream>>>(h5, y);
  conv5_c1_kernel<<<grid(4 * 128 * 128), B, 0, stream>>>(y, w_c1, c1h);

  pack_a1_kernel<<<grid(32 * 13 * 32 * 16), B, 0, stream>>>(w_caps1, a1);
  pack_a2_kernel<<<grid(2 * 400 * 32 * 16), B, 0, stream>>>(w_caps2, a2);

  caps1_wmma_kernel<<<1024, B, 0, stream>>>(c1h, a1, v1h);
  caps2_wmma_kernel<<<512,  B, 0, stream>>>(v1h, a2, s2);

  squash25_kernel<<<grid(4 * 128 * 128), B, 0, stream>>>(s2, v2);
  pixel_conv_add_kernel<<<grid(4 * 128 * 128), B, 0, stream>>>(v2, y, (float*)d_out);
}
